// SelfAttention_32950989095326
// MI455X (gfx1250) — matrix operand order repaired and checked by reading, not compiled, not run on hardware
//
#include <hip/hip_runtime.h>
#include <hip/hip_bf16.h>

// ---------------------------------------------------------------------------
// Self-attention, CDNA5 (gfx1250), bf16 WMMA + double-buffered async
// global->LDS staging.  B=2, L=2048, D=1024, H=16, DH=64.
// ---------------------------------------------------------------------------

typedef __attribute__((ext_vector_type(16))) __bf16 v16bf;
typedef __attribute__((ext_vector_type(8)))  float  v8f;

union Frag16 {           // 32 bytes: one 16x32 bf16 A-frag or 32x16 bf16 B-frag
    v16bf  bf;
    float4 f4[2];
};

__device__ __forceinline__ unsigned short f32_to_bf16(float f) {
    union { float f; unsigned int u; } c; c.f = f;
    unsigned int u = c.u;
    u += 0x7FFFu + ((u >> 16) & 1u);   // round-to-nearest-even
    return (unsigned short)(u >> 16);
}

// Async direct global->LDS copy (16B per lane), tracked by ASYNCcnt.
__device__ __forceinline__ void async_copy_b128(void* lds_ptr, const void* gptr) {
    unsigned lds_off = (unsigned)(unsigned long long)lds_ptr;  // low 32 bits = LDS offset
    asm volatile("global_load_async_to_lds_b128 %0, %1, off"
                 :: "v"(lds_off), "v"(gptr)
                 : "memory");
}
__device__ __forceinline__ void wait_async0() {   // all async done
    asm volatile("s_wait_asynccnt 0x0" ::: "memory");
}
__device__ __forceinline__ void wait_async5() {   // allow 5 in flight (next tile)
    asm volatile("s_wait_asynccnt 0x5" ::: "memory");
}

constexpr int Bc  = 2;
constexpr int Lc  = 2048;
constexpr int Dc  = 1024;
constexpr int Hc  = 16;
constexpr int DHc = 64;
constexpr int Mrows = Bc * Lc;          // 4096
constexpr float NEGM = -10000.0f;

// ---------------------------------------------------------------------------
// fp32 -> bf16 elementwise convert
// ---------------------------------------------------------------------------
__global__ void cvt_f32_bf16(const float* __restrict__ in,
                             unsigned short* __restrict__ out, int n) {
    int i = blockIdx.x * blockDim.x + threadIdx.x;
    if (i < n) out[i] = f32_to_bf16(in[i]);
}

// ---------------------------------------------------------------------------
// C[M,N] = A[M,K] * W[N,K]^T + bias   (A,W bf16; acc f32)
// Block tile 64(M) x 256(N), K-step 32, 256 threads = 8 waves,
// wave tile 32x64 (2x4 C-frags). Double-buffered async global->LDS tiles:
// next K-tile DMA overlaps current tile's WMMAs (ASYNCcnt in-order => wait<=5).
// mode 0: out bf16 head layout  [B,H,L,DH]
// mode 1: out bf16 transposed   [B,H,DH,L]   (for V)
// mode 2: out fp32 row-major    [M,N]        (final projection)
// ---------------------------------------------------------------------------
__global__ __launch_bounds__(256) void gemm_bf16(
    const unsigned short* __restrict__ A,
    const unsigned short* __restrict__ W,
    const float* __restrict__ bias,
    void* __restrict__ outp, int mode)
{
    constexpr int K   = Dc;
    constexpr int N   = Dc;
    constexpr int BK  = 32;
    constexpr int ASZ = 64  * BK;   // elems per A buffer
    constexpr int BSZ = 256 * BK;   // elems per B buffer

    __shared__ unsigned short As[2 * ASZ];   //  8 KB
    __shared__ unsigned short Bs[2 * BSZ];   // 32 KB

    const int tid   = threadIdx.x;
    const int lane  = tid & 31;
    const int wave  = tid >> 5;
    const int ln    = lane & 15;
    const bool lohf = lane < 16;
    const int waveM = wave & 1;               // 2 waves along M (32 rows each)
    const int waveN = wave >> 1;              // 4 waves along N (64 cols each)
    const int m0 = blockIdx.y * 64;
    const int n0 = blockIdx.x * 256;

    v8f acc[2][4];
    for (int i = 0; i < 2; ++i)
        for (int j = 0; j < 4; ++j)
            for (int e = 0; e < 8; ++e) acc[i][j][e] = 0.0f;

    const int arow = tid >> 2;                // 0..63
    const int acol = (tid & 3) * 8;           // 0,8,16,24

    // issue 5 async b128 copies staging the K-tile at k0 into buffer `buf`
    auto stage = [&](int buf, int k0) {
        async_copy_b128(&As[buf * ASZ + arow * BK + acol],
                        &A[(size_t)(m0 + arow) * K + k0 + acol]);
        for (int c = 0; c < 4; ++c)
            async_copy_b128(&Bs[buf * BSZ + tid * BK + c * 8],
                            &W[(size_t)(n0 + tid) * K + k0 + c * 8]);
    };

    stage(0, 0);                              // prologue DMA
    int cur = 0;

    for (int k0 = 0; k0 < K; k0 += BK) {
        if (k0 + BK < K) {
            stage(cur ^ 1, k0 + BK);          // next tile DMA in flight
            wait_async5();                    // current tile landed (in-order cnt)
        } else {
            wait_async0();
        }
        __syncthreads();

        const unsigned short* Asb = &As[cur * ASZ];
        const unsigned short* Bsb = &Bs[cur * BSZ];

        Frag16 afr[2], bfr[4];
        for (int ms = 0; ms < 2; ++ms) {
            const int row  = waveM * 32 + ms * 16 + ln;
            const int koff = lohf ? 0 : 8;    // A-frag: K chunks {0..7|8..15},{16..23|24..31}
            afr[ms].f4[0] = *reinterpret_cast<const float4*>(&Asb[row * BK + koff]);
            afr[ms].f4[1] = *reinterpret_cast<const float4*>(&Asb[row * BK + 16 + koff]);
        }
        for (int ns = 0; ns < 4; ++ns) {
            const int col  = waveN * 64 + ns * 16 + ln;
            const int koff = lohf ? 0 : 8;    // B-frag: K chunks {0..7|8..15},{16..23|24..31}
            bfr[ns].f4[0] = *reinterpret_cast<const float4*>(&Bsb[col * BK + koff]);
            bfr[ns].f4[1] = *reinterpret_cast<const float4*>(&Bsb[col * BK + koff + 16]);
        }
        for (int ms = 0; ms < 2; ++ms)
            for (int ns = 0; ns < 4; ++ns)
                acc[ms][ns] = __builtin_amdgcn_wmma_f32_16x16x32_bf16(
                    false, afr[ms].bf, false, bfr[ns].bf,
                    (short)0, acc[ms][ns], false, false);
        __syncthreads();   // all waves done reading before buffer is re-filled
        cur ^= 1;
    }

    // epilogue: bias + layout-specific scatter
    for (int ms = 0; ms < 2; ++ms)
        for (int ns = 0; ns < 4; ++ns)
            for (int j = 0; j < 8; ++j) {
                const int m = m0 + waveM * 32 + ms * 16 + j + (lohf ? 0 : 8);
                const int n = n0 + waveN * 64 + ns * 16 + ln;
                const float v = acc[ms][ns][j] + bias[n];
                if (mode == 2) {
                    reinterpret_cast<float*>(outp)[(size_t)m * N + n] = v;
                } else {
                    const int b = m / Lc, l = m % Lc;
                    const int h = n / DHc, dh = n % DHc;
                    unsigned short* o = reinterpret_cast<unsigned short*>(outp);
                    if (mode == 0)
                        o[((size_t)(b * Hc + h) * Lc + l) * DHc + dh] = f32_to_bf16(v);
                    else
                        o[((size_t)(b * Hc + h) * DHc + dh) * Lc + l] = f32_to_bf16(v);
                }
            }
}

// ---------------------------------------------------------------------------
// Fused causal flash attention, one (b,h,64-query-block) per workgroup.
// 4 waves x 16 query rows; streams 32-key blocks with online softmax.
// Q,K: bf16 [B,H,L,DH]; V: bf16 [B,H,DH,L] (pre-transposed); out bf16 [B,L,D].
// ---------------------------------------------------------------------------
__global__ __launch_bounds__(128) void flash_attn(
    const unsigned short* __restrict__ Qh,
    const unsigned short* __restrict__ Kh,
    const unsigned short* __restrict__ Vt,
    const int* __restrict__ mk,
    unsigned short* __restrict__ Yb)
{
    __shared__ unsigned short Plds[4 * 16 * 32];   // per-wave P relayout buffer

    const int tid  = threadIdx.x;
    const int lane = tid & 31;
    const int wave = tid >> 5;
    const int ln   = lane & 15;
    const bool lohf = lane < 16;

    const int qb0 = blockIdx.x * 64;
    const int h   = blockIdx.y;
    const int b   = blockIdx.z;

    const unsigned short* Qp = Qh + (size_t)((b * Hc + h) * Lc) * DHc;
    const unsigned short* Kp = Kh + (size_t)((b * Hc + h) * Lc) * DHc;
    const unsigned short* Vp = Vt + (size_t)((b * Hc + h) * DHc) * Lc;
    const int* mp = mk + b * Lc;
    unsigned short* pbase = &Plds[wave * 16 * 32];

    const int qrow0 = qb0 + wave * 16;

    // resident Q A-frags (DH = 2 x K32)
    Frag16 qf[2];
    for (int d = 0; d < 2; ++d) {
        const int q  = qrow0 + ln;
        const int ko = d * 32 + (lohf ? 0 : 8);
        qf[d].f4[0] = *reinterpret_cast<const float4*>(&Qp[q * DHc + ko]);
        qf[d].f4[1] = *reinterpret_cast<const float4*>(&Qp[q * DHc + ko + 16]);
    }

    v8f o[4];
    float mrow[8], lrow[8];
    for (int n = 0; n < 4; ++n)
        for (int e = 0; e < 8; ++e) o[n][e] = 0.0f;
    for (int j = 0; j < 8; ++j) { mrow[j] = -1e30f; lrow[j] = 0.0f; }

    const int nkb = (qb0 >> 5) + 2;   // causal: keys <= qb0+63 only

    for (int kb = 0; kb < nkb; ++kb) {
        const int kbase = kb * 32;

        // prefetch next key block (K rows + V rows) into near caches
        if (kb + 1 < nkb) {
            __builtin_prefetch(&Kp[(kbase + 32 + lane) * DHc], 0, 3);
            __builtin_prefetch(&Vp[(lane * 2) * Lc + kbase + 32], 0, 3);
        }

        // S = Q K^T  (two 16-key subtiles)
        v8f c[2];
        for (int s = 0; s < 2; ++s) {
            Frag16 kf0, kf1;
            const int col = kbase + s * 16 + ln;
            const int ko  = lohf ? 0 : 8;
            kf0.f4[0] = *reinterpret_cast<const float4*>(&Kp[col * DHc + ko]);
            kf0.f4[1] = *reinterpret_cast<const float4*>(&Kp[col * DHc + ko + 16]);
            kf1.f4[0] = *reinterpret_cast<const float4*>(&Kp[col * DHc + 32 + ko]);
            kf1.f4[1] = *reinterpret_cast<const float4*>(&Kp[col * DHc + 32 + ko + 16]);
            v8f z; for (int e = 0; e < 8; ++e) z[e] = 0.0f;
            z = __builtin_amdgcn_wmma_f32_16x16x32_bf16(false, qf[0].bf, false, kf0.bf,
                                                        (short)0, z, false, false);
            z = __builtin_amdgcn_wmma_f32_16x16x32_bf16(false, qf[1].bf, false, kf1.bf,
                                                        (short)0, z, false, false);
            c[s] = z;
        }

        // scale + causal + pad mask
        float p[2][8];
        for (int s = 0; s < 2; ++s)
            for (int j = 0; j < 8; ++j) {
                const int key = kbase + s * 16 + ln;
                const int q   = qrow0 + j + (lohf ? 0 : 8);
                float v = c[s][j] * 0.125f;           // 1/sqrt(64)
                if (key > q) v = NEGM;
                v += (1.0f - (float)mp[key]) * NEGM;
                p[s][j] = v;
            }

        // online softmax (row reductions across each 16-lane half)
        for (int j = 0; j < 8; ++j) {
            float t = fmaxf(p[0][j], p[1][j]);
            for (int msk = 1; msk < 16; msk <<= 1)
                t = fmaxf(t, __shfl_xor(t, msk, 16));
            const float mn   = fmaxf(mrow[j], t);
            const float corr = __expf(mrow[j] - mn);
            mrow[j] = mn;
            p[0][j] = __expf(p[0][j] - mn);
            p[1][j] = __expf(p[1][j] - mn);
            float rs = p[0][j] + p[1][j];
            for (int msk = 1; msk < 16; msk <<= 1)
                rs += __shfl_xor(rs, msk, 16);
            lrow[j] = lrow[j] * corr + rs;
            for (int n = 0; n < 4; ++n) o[n][j] *= corr;
        }

        // relayout P: C-frag -> LDS row-major -> A-frag
        for (int s = 0; s < 2; ++s)
            for (int j = 0; j < 8; ++j)
                pbase[(j + (lohf ? 0 : 8)) * 32 + s * 16 + ln] = f32_to_bf16(p[s][j]);
        __syncthreads();   // nkb uniform across block; orders DS store->load

        Frag16 pf;
        {
            const int ko = lohf ? 0 : 8;
            pf.f4[0] = *reinterpret_cast<const float4*>(&pbase[ln * 32 + ko]);
            pf.f4[1] = *reinterpret_cast<const float4*>(&pbase[ln * 32 + 16 + ko]);
        }

        // O += P V   (V pre-transposed: contiguous along keys)
        for (int n = 0; n < 4; ++n) {
            Frag16 vf;
            const int dcol = n * 16 + ln;
            const int kk   = kbase + (lohf ? 0 : 8);
            vf.f4[0] = *reinterpret_cast<const float4*>(&Vp[dcol * Lc + kk]);
            vf.f4[1] = *reinterpret_cast<const float4*>(&Vp[dcol * Lc + kk + 16]);
            o[n] = __builtin_amdgcn_wmma_f32_16x16x32_bf16(false, pf.bf, false, vf.bf,
                                                           (short)0, o[n], false, false);
        }
        __syncthreads();
    }

    // normalize and write merged-head bf16 output [B, L, D]
    for (int n = 0; n < 4; ++n)
        for (int j = 0; j < 8; ++j) {
            const int q  = qrow0 + j + (lohf ? 0 : 8);
            const int dh = n * 16 + ln;
            const float v = o[n][j] * (1.0f / lrow[j]);
            Yb[((size_t)(b * Lc + q)) * Dc + h * DHc + dh] = f32_to_bf16(v);
        }
}

// ---------------------------------------------------------------------------
// launcher
// ---------------------------------------------------------------------------
extern "C" void kernel_launch(void* const* d_in, const int* in_sizes, int n_in,
                              void* d_out, int out_size, void* d_ws, size_t ws_size,
                              hipStream_t stream) {
    (void)in_sizes; (void)n_in; (void)out_size; (void)ws_size;

    const float* x  = (const float*)d_in[0];
    const int*   mk = (const int*)  d_in[1];
    const float* Wq = (const float*)d_in[2];
    const float* bq = (const float*)d_in[3];
    const float* Wk = (const float*)d_in[4];
    const float* bk = (const float*)d_in[5];
    const float* Wv = (const float*)d_in[6];
    const float* bv = (const float*)d_in[7];
    const float* Wp = (const float*)d_in[8];
    const float* bp = (const float*)d_in[9];
    float* out = (float*)d_out;

    char* ws = (char*)d_ws;
    size_t off = 0;
    auto carve = [&](size_t bytes) { char* p = ws + off; off += bytes; return p; };

    unsigned short* xb  = (unsigned short*)carve((size_t)Mrows * Dc * 2);
    unsigned short* wqb = (unsigned short*)carve((size_t)Dc * Dc * 2);
    unsigned short* wkb = (unsigned short*)carve((size_t)Dc * Dc * 2);
    unsigned short* wvb = (unsigned short*)carve((size_t)Dc * Dc * 2);
    unsigned short* wpb = (unsigned short*)carve((size_t)Dc * Dc * 2);
    unsigned short* Qh  = (unsigned short*)carve((size_t)Mrows * Dc * 2);
    unsigned short* Kh  = (unsigned short*)carve((size_t)Mrows * Dc * 2);
    unsigned short* Vt  = (unsigned short*)carve((size_t)Mrows * Dc * 2);
    unsigned short* yb  = (unsigned short*)carve((size_t)Mrows * Dc * 2);

    // 1) bf16 conversions
    {
        const int nx = Mrows * Dc;
        cvt_f32_bf16<<<(nx + 255) / 256, 256, 0, stream>>>(x, xb, nx);
        const int nw = Dc * Dc;
        cvt_f32_bf16<<<(nw + 255) / 256, 256, 0, stream>>>(Wq, wqb, nw);
        cvt_f32_bf16<<<(nw + 255) / 256, 256, 0, stream>>>(Wk, wkb, nw);
        cvt_f32_bf16<<<(nw + 255) / 256, 256, 0, stream>>>(Wv, wvb, nw);
        cvt_f32_bf16<<<(nw + 255) / 256, 256, 0, stream>>>(Wp, wpb, nw);
    }

    // 2) QKV projections (WMMA GEMM, double-buffered async-LDS staging)
    dim3 gg(Dc / 256, Mrows / 64);      // (4, 64)
    gemm_bf16<<<gg, 256, 0, stream>>>(xb, wqb, bq, (void*)Qh, 0);
    gemm_bf16<<<gg, 256, 0, stream>>>(xb, wkb, bk, (void*)Kh, 0);
    gemm_bf16<<<gg, 256, 0, stream>>>(xb, wvb, bv, (void*)Vt, 1);

    // 3) fused causal flash attention
    dim3 ga(Lc / 64, Hc, Bc);           // (32, 16, 2)
    flash_attn<<<ga, 128, 0, stream>>>(Qh, Kh, Vt, mk, yb);

    // 4) output projection -> fp32 d_out
    gemm_bf16<<<gg, 256, 0, stream>>>(yb, wpb, bp, (void*)out, 2);
}
